// LearnableSeedCLF_3547642986554
// MI455X (gfx1250) — matrix operand order repaired and checked by reading, not compiled, not run on hardware
//
#include <hip/hip_runtime.h>

typedef __attribute__((ext_vector_type(16))) _Float16 v16h;
typedef __attribute__((ext_vector_type(8)))  _Float16 v8h;
typedef __attribute__((ext_vector_type(8)))  float    v8f;

#define BATCH 2048
#define SEQ   200
#define VOCAB 32000
#define NCLS  9
#define CPAD  16   // classes padded to 16 for the WMMA tile

// ---------------------------------------------------------------------------
// Prep: W[9][32000] f32 (row-major, class-major) -> Wt16[32000][16] f16,
// classes 9..15 zero-padded. One thread per vocab entry; two 16B stores.
// ---------------------------------------------------------------------------
__global__ __launch_bounds__(256) void bow_pack_wt16(const float* __restrict__ W,
                                                     _Float16* __restrict__ Wt) {
    int v = blockIdx.x * blockDim.x + threadIdx.x;
    if (v >= VOCAB) return;
    _Float16 vals[CPAD];
#pragma unroll
    for (int c = 0; c < CPAD; ++c) vals[c] = (_Float16)0.0f;
#pragma unroll
    for (int c = 0; c < NCLS; ++c) vals[c] = (_Float16)W[c * VOCAB + v];
    v8h lo, hi;
#pragma unroll
    for (int j = 0; j < 8; ++j) { lo[j] = vals[j]; hi[j] = vals[8 + j]; }
    v8h* dst = (v8h*)(Wt + (size_t)v * CPAD);   // 32B-aligned
    dst[0] = lo;
    dst[1] = hi;
}

// ---------------------------------------------------------------------------
// Main: one wave32 per 16-row batch tile. 100 x v_wmma_f32_16x16x32_f16 over
// the 200-token loop (2 tokens x 16 class slots per WMMA) in TWO independent
// accumulator chains, f32 accumulate, then LDS transpose + 9-way softmax.
//
// A layout (16-bit 16x32, ISA 7.12.2): lane L holds row M=L&15;
//   VGPRs 0-3 -> K = clsBase..clsBase+7          (token 0, clsBase=(L<16)?0:8)
//   VGPRs 4-7 -> K = 16+clsBase..16+clsBase+7    (token 1)
// B layout (32x16): lane n holds column N=n&15, same K order as A (AMD table
// order)  =>  bsel[j] = ((j&7) + clsBase == n) encodes B[k][n]=(k%16==n).
// D layout: VGPR r, lanes 0-15 -> (M=r, N=lane); lanes 16-31 -> (M=8+r).
//
// Gathers use uniform-base + 32-bit byte offset (id*32 + clsByte < 1 MB) so
// the backend can emit global_load_b128 saddr form with a single 32-bit
// v_lshl_add per address instead of 64-bit shift/add chains.
// ---------------------------------------------------------------------------
__global__ __launch_bounds__(32) void bow_wmma_softmax(const int* __restrict__ ids,
                                                       const _Float16* __restrict__ Wt,
                                                       const float* __restrict__ bias,
                                                       float* __restrict__ out) {
    const int      lane    = threadIdx.x;            // 0..31
    const int      tile    = blockIdx.x;             // 0..127
    const int      mrow    = tile * 16 + (lane & 15);
    const uint32_t cByte   = (lane < 16) ? 0u : 16u; // clsBase * sizeof(f16)

    __shared__ float logits[16][CPAD];

    const char* __restrict__ WtB = (const char*)Wt;        // uniform 64-bit base
    const int4* idRow4 = (const int4*)(ids + (size_t)mrow * SEQ);  // 800B rows -> 16B aligned

    // B: class-selection matrix, two nonzero f16 per column (k=n, k=16+n).
    const int n = lane & 15;
    v16h bsel;
#pragma unroll
    for (int j = 0; j < 16; ++j) bsel[j] = ((j & 7) + ((lane < 16) ? 0 : 8) == n) ? (_Float16)1.0f : (_Float16)0.0f;

    v8f acc0 = {};
    v8f acc1 = {};
    for (int it = 0; it < SEQ / 4; ++it) {   // 50 iters, 4 tokens, 2 WMMAs each
        int4 q = idRow4[it];                                   // global_load_b128 (ids)
        uint32_t o0 = ((uint32_t)q.x << 5) + cByte;
        uint32_t o1 = ((uint32_t)q.y << 5) + cByte;
        uint32_t o2 = ((uint32_t)q.z << 5) + cByte;
        uint32_t o3 = ((uint32_t)q.w << 5) + cByte;
        v8h a0 = *(const v8h*)(WtB + o0);                      // saddr gather b128
        v8h a1 = *(const v8h*)(WtB + o1);
        v8h a2 = *(const v8h*)(WtB + o2);
        v8h a3 = *(const v8h*)(WtB + o3);
        v16h a01 = __builtin_shufflevector(a0, a1, 0, 1, 2, 3, 4, 5, 6, 7,
                                                   8, 9, 10, 11, 12, 13, 14, 15);
        v16h a23 = __builtin_shufflevector(a2, a3, 0, 1, 2, 3, 4, 5, 6, 7,
                                                   8, 9, 10, 11, 12, 13, 14, 15);
        acc0 = __builtin_amdgcn_wmma_f32_16x16x32_f16(
            false, a01, false, bsel, (short)0, acc0, false, false);
        acc1 = __builtin_amdgcn_wmma_f32_16x16x32_f16(
            false, a23, false, bsel, (short)0, acc1, false, false);
    }
    v8f acc = acc0 + acc1;

    // Scatter D tile (+bias on real classes) into LDS: lane n holds class n.
    const int   mBase = (lane < 16) ? 0 : 8;
    const float bn    = (n < NCLS) ? bias[n] : 0.0f;
#pragma unroll
    for (int r = 0; r < 8; ++r) logits[mBase + r][n] = acc[r] + bn;
    __syncthreads();

    // 9-way softmax: one lane per batch row of the tile.
    if (lane < 16) {
        float mx = logits[lane][0];
#pragma unroll
        for (int c = 1; c < NCLS; ++c) mx = fmaxf(mx, logits[lane][c]);
        float e[NCLS], s = 0.0f;
#pragma unroll
        for (int c = 0; c < NCLS; ++c) { e[c] = __expf(logits[lane][c] - mx); s += e[c]; }
        const float inv = 1.0f / s;
        float* o = out + (size_t)(tile * 16 + lane) * NCLS;
#pragma unroll
        for (int c = 0; c < NCLS; ++c) o[c] = e[c] * inv;
    }
}

// ---------------------------------------------------------------------------
// Fallback (only if workspace is too small for Wt16): pure gather-sum.
// ---------------------------------------------------------------------------
__global__ __launch_bounds__(256) void bow_scalar(const int* __restrict__ ids,
                                                  const float* __restrict__ W,
                                                  const float* __restrict__ bias,
                                                  float* __restrict__ out) {
    int m = blockIdx.x * blockDim.x + threadIdx.x;
    if (m >= BATCH) return;
    float acc[NCLS];
#pragma unroll
    for (int c = 0; c < NCLS; ++c) acc[c] = bias[c];
    const int* row = ids + (size_t)m * SEQ;
    for (int t = 0; t < SEQ; ++t) {
        int id = row[t];
#pragma unroll
        for (int c = 0; c < NCLS; ++c) acc[c] += W[c * VOCAB + id];
    }
    float mx = acc[0];
#pragma unroll
    for (int c = 1; c < NCLS; ++c) mx = fmaxf(mx, acc[c]);
    float e[NCLS], s = 0.0f;
#pragma unroll
    for (int c = 0; c < NCLS; ++c) { e[c] = __expf(acc[c] - mx); s += e[c]; }
    float inv = 1.0f / s;
#pragma unroll
    for (int c = 0; c < NCLS; ++c) out[(size_t)m * NCLS + c] = e[c] * inv;
}

extern "C" void kernel_launch(void* const* d_in, const int* in_sizes, int n_in,
                              void* d_out, int out_size, void* d_ws, size_t ws_size,
                              hipStream_t stream) {
    const int*   ids  = (const int*)d_in[0];    // [2048][200] int32
    const float* W    = (const float*)d_in[1];  // [9][32000]  f32
    const float* bias = (const float*)d_in[2];  // [9]         f32
    float*       out  = (float*)d_out;          // [2048][9]   f32

    const size_t wt_bytes = (size_t)VOCAB * CPAD * sizeof(_Float16);  // 1,024,000 B
    if (ws_size >= wt_bytes) {
        _Float16* Wt = (_Float16*)d_ws;
        bow_pack_wt16<<<(VOCAB + 255) / 256, 256, 0, stream>>>(W, Wt);
        bow_wmma_softmax<<<BATCH / 16, 32, 0, stream>>>(ids, Wt, bias, out);
    } else {
        bow_scalar<<<(BATCH + 255) / 256, 256, 0, stream>>>(ids, W, bias, out);
    }
}
